// SuperPoint_13640816132776
// MI455X (gfx1250) — matrix operand order repaired and checked by reading, not compiled, not run on hardware
//
#include <hip/hip_runtime.h>
#include <hip/hip_bf16.h>
#include <stdint.h>

typedef _Float16 half_t;
typedef __attribute__((ext_vector_type(16))) _Float16 v16h;
typedef __attribute__((ext_vector_type(8)))  _Float16 v8h;
typedef __attribute__((ext_vector_type(8)))  float    v8f;
typedef __attribute__((ext_vector_type(4)))  unsigned int v4u;
typedef __attribute__((ext_vector_type(4)))  int      v4i;
typedef __attribute__((ext_vector_type(8)))  int      v8i;

#define BN_EPS 0.001f

#if defined(__HIP_DEVICE_COMPILE__) && __has_builtin(__builtin_amdgcn_tensor_load_to_lds) && __has_builtin(__builtin_amdgcn_s_wait_tensorcnt)
#define HAVE_TDM 1
#else
#define HAVE_TDM 0
#endif

// ---------------------------------------------------------------------------
// One-time weight prep: OIHW f32 -> f16 GEMM layout [CoutPad][Kpad],
// k = tap*Cin + c, zero padded.  All div/mod happens here, once.
// ---------------------------------------------------------------------------
__global__ void k_prep_w(const float* __restrict__ wgt, half_t* __restrict__ out,
                         int Cin, int Cout, int ksize, int CoutPad, int Kpad)
{
    long total = (long)CoutPad * Kpad;
    long i = (long)blockIdx.x * blockDim.x + threadIdx.x;
    if (i >= total) return;
    int k = (int)(i % Kpad);
    int m = (int)(i / Kpad);
    int Ktot = ksize * ksize * Cin;
    float v = 0.f;
    if (m < Cout && k < Ktot) {
        int tap = k / Cin, c = k - tap * Cin;
        int ky = tap / ksize, kx = tap - ky * ksize;
        v = wgt[((long)(m * Cin + c) * ksize + ky) * ksize + kx];
    }
    out[i] = (half_t)v;
}

// ---------------------------------------------------------------------------
// Implicit-GEMM conv (ksize 1 or 3) + ReLU + BN fused, channels-last f16.
// Weight tile (16 x Kpad f16) staged into LDS by the Tensor Data Mover
// (wave 0 issues tensor_load_to_lds, s_wait_tensorcnt, block sync);
// cooperative v8h copy fallback when the builtin is unavailable.
// Each wave computes a 16 out-channel x 32 pixel tile: one shared A
// fragment feeds two WMMA f32_16x16x32_f16 accumulators per K-step.
// ---------------------------------------------------------------------------
__global__ __launch_bounds__(256)
void k_conv_wmma(const half_t* __restrict__ in, const half_t* __restrict__ wprep,
                 const float* __restrict__ scale, const float* __restrict__ bias,
                 const float* __restrict__ mean, const float* __restrict__ var,
                 half_t* __restrict__ out,
                 int Nimg, int H, int W, int Cin, int Cout, int ksize, int relu, int Kpad)
{
    __shared__ __align__(32) half_t lw[16 * 1184];   // max K = 1152

    const int tid   = threadIdx.x;
    const int mtile = blockIdx.x;
    const int nW    = 16 * Kpad;                      // halves in weight tile
    const half_t* wsrc = wprep + (long)mtile * nW;

#if HAVE_TDM
    if (tid < 32) {                                   // wave 0 only (uniform)
        unsigned long long ga = (unsigned long long)(uintptr_t)wsrc;
        unsigned lds_off = (unsigned)(uintptr_t)lw;   // low 32 bits = LDS offset
        unsigned n = (unsigned)nW;                    // elements of 2 bytes
        v4u g0;
        g0[0] = 1u;                                   // count=1, user mode
        g0[1] = lds_off;                              // lds_addr
        g0[2] = (unsigned)ga;                         // global_addr[31:0]
        g0[3] = (unsigned)((ga >> 32) & 0x01FFFFFFu) | 0x80000000u; // [56:32] | type=2
        v8i g1;
        g1[0] = 0x00010000;                           // wg_mask=0, data_size=2B
        g1[1] = (int)((n & 0xFFFFu) << 16);           // tensor_dim0[15:0]
        g1[2] = (int)((n >> 16) | (1u << 16));        // tensor_dim0[31:16], tensor_dim1=1
        g1[3] = (int)((n & 0xFFFFu) << 16);           // tile_dim0 (n <= 18432 fits 16b)
        g1[4] = 1;                                    // tile_dim1=1, tile_dim2=0
        g1[5] = (int)n;                               // tensor_dim0_stride[31:0]
        g1[6] = 0;                                    // stride0 hi, stride1 lo
        g1[7] = 0;                                    // stride1 hi
        v4i gz; gz[0] = 0; gz[1] = 0; gz[2] = 0; gz[3] = 0;
#if defined(__clang_major__) && __clang_major__ >= 23
        v8i gz8; gz8[0]=0; gz8[1]=0; gz8[2]=0; gz8[3]=0; gz8[4]=0; gz8[5]=0; gz8[6]=0; gz8[7]=0;
        __builtin_amdgcn_tensor_load_to_lds(g0, g1, gz, gz, gz8, 0);
#else
        __builtin_amdgcn_tensor_load_to_lds(g0, g1, gz, gz, 0);
#endif
        __builtin_amdgcn_s_wait_tensorcnt(0);
    }
#else
    for (int i = tid * 8; i < nW; i += 256 * 8)
        *(v8h*)(lw + i) = *(const v8h*)(wsrc + i);
#endif
    __syncthreads();

    const int wave = tid >> 5;
    const int lane = tid & 31;
    const int tilesPerRow = W >> 5;                  // 32 pixels per wave tile
    const long tileId = (long)blockIdx.y * 8 + wave;
    const long totalTiles = (long)Nimg * H * tilesPerRow;
    if (tileId >= totalTiles) return;                // wave-uniform

    int t   = (int)tileId;
    int x0  = (t % tilesPerRow) << 5;  t /= tilesPerRow;
    int y   = t % H;
    int img = t / H;

    const int n    = lane & 15;              // pixel column (B/C/D layout)
    const int x    = x0 + n;                 // fragment 0; fragment 1 = x+16
    const int upr  = lane >> 4;
    const int aof0 = upr ? 16 : 0;           // A K-runs {0..15} / {16..31}
    const int aof1 = upr ? 24 : 8;
    const half_t* lwm = lw + (lane & 15) * Kpad;
    const int pad = (ksize - 1) >> 1;
    const int kbo = upr << 4;                // B: lanes 16-31 hold K+16

    v8f acc0 = {0.f, 0.f, 0.f, 0.f, 0.f, 0.f, 0.f, 0.f};
    v8f acc1 = {0.f, 0.f, 0.f, 0.f, 0.f, 0.f, 0.f, 0.f};

    if ((Cin & 31) == 0) {
        int tap = 0;
        for (int ky = -pad; ky <= pad; ++ky) {
            const int yy = y + ky;
            for (int kx = -pad; kx <= pad; ++kx, ++tap) {
                const int xx0 = x + kx;
                const int xx1 = x + 16 + kx;
                const bool yok  = (yy >= 0) & (yy < H);
                const bool inb0 = yok & (xx0 >= 0) & (xx0 < W);
                const bool inb1 = yok & (xx1 >= 0) & (xx1 < W);
                const half_t* src0 = in + (((long)img * H + yy) * W + xx0) * Cin + kbo;
                const half_t* src1 = src0 + 16 * Cin;
                const half_t* wk   = lwm + tap * Cin;
#pragma unroll 2
                for (int c0 = 0; c0 < Cin; c0 += 32) {
                    v8h alo = *(const v8h*)(wk + c0 + aof0);
                    v8h ahi = *(const v8h*)(wk + c0 + aof1);
                    v16h a;
#pragma unroll
                    for (int j = 0; j < 8; ++j) { a[j] = alo[j]; a[j + 8] = ahi[j]; }
                    v16h b0, b1;
                    if (inb0) {
                        b0 = *(const v16h*)(src0 + c0);
                    } else {
#pragma unroll
                        for (int j = 0; j < 16; ++j) b0[j] = (half_t)0.f;
                    }
                    if (inb1) {
                        b1 = *(const v16h*)(src1 + c0);
                    } else {
#pragma unroll
                        for (int j = 0; j < 16; ++j) b1[j] = (half_t)0.f;
                    }
                    acc0 = __builtin_amdgcn_wmma_f32_16x16x32_f16(
                               false, a, false, b0, (short)0, acc0, false, false);
                    acc1 = __builtin_amdgcn_wmma_f32_16x16x32_f16(
                               false, a, false, b1, (short)0, acc1, false, false);
                }
            }
        }
    } else {
        // generic path (only the 1-channel first conv): K = ksize*ksize*Cin
        const int Ktot = ksize * ksize * Cin;
        for (int k0 = 0; k0 < Kpad; k0 += 32) {
            v8h alo = *(const v8h*)(lwm + k0 + aof0);
            v8h ahi = *(const v8h*)(lwm + k0 + aof1);
            v16h a;
#pragma unroll
            for (int j = 0; j < 8; ++j) { a[j] = alo[j]; a[j + 8] = ahi[j]; }
            v16h b0, b1;
            const int kbase = k0 + kbo;
#pragma unroll
            for (int j = 0; j < 16; ++j) {
                int k = kbase + j;
                half_t v0 = (half_t)0.f, v1 = (half_t)0.f;
                if (k < Ktot) {
                    int tap = k / Cin, c = k - tap * Cin;
                    int ky = tap / ksize - pad, kx = tap % ksize - pad;
                    int yy = y + ky;
                    if (yy >= 0 && yy < H) {
                        int xa = x + kx, xb = x + 16 + kx;
                        const half_t* row = in + (((long)img * H + yy) * W) * Cin + c;
                        if (xa >= 0 && xa < W) v0 = row[(long)xa * Cin];
                        if (xb >= 0 && xb < W) v1 = row[(long)xb * Cin];
                    }
                }
                b0[j] = v0;
                b1[j] = v1;
            }
            acc0 = __builtin_amdgcn_wmma_f32_16x16x32_f16(
                       false, a, false, b0, (short)0, acc0, false, false);
            acc1 = __builtin_amdgcn_wmma_f32_16x16x32_f16(
                       false, a, false, b1, (short)0, acc1, false, false);
        }
    }

    // ---- epilogue: ReLU -> BN -> f16 store (two pixel fragments) ----
    const int mb = (mtile << 4) + (upr ? 8 : 0);
    float bnmul[8], bnadd[8];
#pragma unroll
    for (int r = 0; r < 8; ++r) {
        int m = mb + r;
        if (m < Cout) {
            float inv = rsqrtf(var[m] + BN_EPS) * scale[m];
            bnmul[r] = inv;
            bnadd[r] = bias[m] - mean[m] * inv;
        } else {
            bnmul[r] = 0.f; bnadd[r] = 0.f;
        }
    }
    half_t* op0 = out + (((long)img * H + y) * W + x) * Cout + mb;
    half_t* op1 = op0 + 16 * Cout;
    if ((Cout & 15) == 0) {
        v8h pk0, pk1;
#pragma unroll
        for (int r = 0; r < 8; ++r) {
            float v0 = acc0[r], v1 = acc1[r];
            if (relu) { v0 = v0 > 0.f ? v0 : 0.f; v1 = v1 > 0.f ? v1 : 0.f; }
            pk0[r] = (half_t)(v0 * bnmul[r] + bnadd[r]);
            pk1[r] = (half_t)(v1 * bnmul[r] + bnadd[r]);
        }
        *(v8h*)op0 = pk0;
        *(v8h*)op1 = pk1;
    } else {
#pragma unroll
        for (int r = 0; r < 8; ++r) {
            if (mb + r < Cout) {
                float v0 = acc0[r], v1 = acc1[r];
                if (relu) { v0 = v0 > 0.f ? v0 : 0.f; v1 = v1 > 0.f ? v1 : 0.f; }
                op0[r] = (half_t)(v0 * bnmul[r] + bnadd[r]);
                op1[r] = (half_t)(v1 * bnmul[r] + bnadd[r]);
            }
        }
    }
}

// ---------------------------------------------------------------------------
__global__ void k_cvt_img(const float* __restrict__ in, half_t* __restrict__ out, long total)
{
    long i = (long)blockIdx.x * blockDim.x + threadIdx.x;
    if (i < total) out[i] = (half_t)in[i];
}

__global__ void k_maxpool2(const half_t* __restrict__ in, half_t* __restrict__ out,
                           int Nimg, int H, int W, int C)
{
    int Ho = H >> 1, Wo = W >> 1;
    long total = (long)Nimg * Ho * Wo * C;
    long i = (long)blockIdx.x * blockDim.x + threadIdx.x;
    if (i >= total) return;
    int c = i % C; long t = i / C;
    int xo = t % Wo; t /= Wo;
    int yo = t % Ho; int nimg = (int)(t / Ho);
    const half_t* p = in + (((long)nimg * H + (yo << 1)) * W + (xo << 1)) * C + c;
    float a = (float)p[0], b = (float)p[C];
    float d = (float)p[(long)W * C], e = (float)p[(long)W * C + C];
    out[i] = (half_t)fmaxf(fmaxf(a, b), fmaxf(d, e));
}

__global__ void k_softmax_shuffle(const half_t* __restrict__ s, float* __restrict__ heat, int Nimg)
{
    int total = Nimg * 64 * 64;
    int i = blockIdx.x * blockDim.x + threadIdx.x;
    if (i >= total) return;
    int xw = i % 64; int t = i / 64;
    int yh = t % 64; int nimg = t / 64;
    const half_t* p = s + (long)i * 65;
    float mx = -1e30f;
    for (int c = 0; c < 65; ++c) mx = fmaxf(mx, (float)p[c]);
    float e[64], sum = 0.f;
    for (int c = 0; c < 65; ++c) {
        float v = __expf((float)p[c] - mx);
        if (c < 64) e[c] = v;
        sum += v;
    }
    float inv = 1.f / sum;
    for (int c = 0; c < 64; ++c) {
        int s1 = c >> 3, s2 = c & 7;
        heat[(((long)nimg * 512) + yh * 8 + s1) * 512 + xw * 8 + s2] = e[c] * inv;
    }
}

__global__ void k_pool9(const float* __restrict__ in, float* __restrict__ out, int Nimg)
{
    long total = (long)Nimg * 512 * 512;
    long i = (long)blockIdx.x * blockDim.x + threadIdx.x;
    if (i >= total) return;
    int x = i % 512; long t = i / 512;
    int y = t % 512; int nimg = (int)(t / 512);
    const float* p = in + (long)nimg * 512 * 512;
    float m = -3.4e38f;
    for (int dy = -4; dy <= 4; ++dy) {
        int yy = y + dy;
        if (yy < 0 || yy >= 512) continue;
        for (int dx = -4; dx <= 4; ++dx) {
            int xx = x + dx;
            if (xx < 0 || xx >= 512) continue;
            m = fmaxf(m, p[(long)yy * 512 + xx]);
        }
    }
    out[i] = m;
}

__global__ void k_eq(const float* __restrict__ a, const float* __restrict__ b,
                     float* __restrict__ out, long total)
{
    long i = (long)blockIdx.x * blockDim.x + threadIdx.x;
    if (i < total) out[i] = (a[i] == b[i]) ? 1.f : 0.f;
}

__global__ void k_supp(const float* __restrict__ maskpool, const float* __restrict__ heat,
                       float* __restrict__ supp_scores, float* __restrict__ supp_mask, long total)
{
    long i = (long)blockIdx.x * blockDim.x + threadIdx.x;
    if (i >= total) return;
    float sm = (maskpool[i] > 0.f) ? 1.f : 0.f;
    supp_mask[i]   = sm;
    supp_scores[i] = (sm != 0.f) ? 0.f : heat[i];
}

__global__ void k_upd(float* __restrict__ mm, const float* __restrict__ ss,
                      const float* __restrict__ ssp, const float* __restrict__ supp, long total)
{
    long i = (long)blockIdx.x * blockDim.x + threadIdx.x;
    if (i >= total) return;
    bool new_max = (ss[i] == ssp[i]) && (supp[i] == 0.f);
    mm[i] = ((mm[i] != 0.f) || new_max) ? 1.f : 0.f;
}

__global__ void k_fin(const float* __restrict__ mm, const float* __restrict__ heat,
                      float* __restrict__ out, int Nimg)
{
    long total = (long)Nimg * 512 * 512;
    long i = (long)blockIdx.x * blockDim.x + threadIdx.x;
    if (i >= total) return;
    int x = i % 512; long t = i / 512;
    int y = t % 512;
    float v = (mm[i] != 0.f) ? heat[i] : 0.f;
    if (x < 4 || y < 4 || x >= 508 || y >= 508) v = -1.f;
    out[i] = v;
}

__global__ __launch_bounds__(1024)
void k_topk(const float* __restrict__ sc, float* __restrict__ kpts_out,
            float* __restrict__ vals_out)
{
    __shared__ unsigned long long cand[4096];
    const int img = blockIdx.x;
    const int tid = threadIdx.x;
    const float* s = sc + (long)img * 512 * 512;

    unsigned long long best[4] = {0ull, 0ull, 0ull, 0ull};
    for (int i = tid; i < 512 * 512; i += 1024) {
        unsigned u = __float_as_uint(s[i]);
        u = (u & 0x80000000u) ? ~u : (u | 0x80000000u);
        unsigned long long key = ((unsigned long long)u << 32) | (unsigned)i;
#pragma unroll
        for (int r = 0; r < 4; ++r) {
            if (key > best[r]) { unsigned long long tmp = best[r]; best[r] = key; key = tmp; }
        }
    }
#pragma unroll
    for (int r = 0; r < 4; ++r) cand[tid * 4 + r] = best[r];
    __syncthreads();

    for (int k = 2; k <= 4096; k <<= 1) {
        for (int j = k >> 1; j > 0; j >>= 1) {
            for (int i = tid; i < 4096; i += 1024) {
                int ixj = i ^ j;
                if (ixj > i) {
                    bool up = (i & k) == 0;
                    unsigned long long a = cand[i], b = cand[ixj];
                    if (up ? (a < b) : (a > b)) { cand[i] = b; cand[ixj] = a; }
                }
            }
            __syncthreads();
        }
    }

    unsigned long long kk = cand[tid];
    unsigned idx = (unsigned)(kk & 0xffffffffu);
    long o = (long)img * 1024 + tid;
    kpts_out[o * 2 + 0] = (float)(idx % 512);
    kpts_out[o * 2 + 1] = (float)(idx / 512);
    vals_out[o] = s[idx];
}

__global__ __launch_bounds__(256)
void k_descnorm(half_t* __restrict__ d)
{
    __shared__ float red[256];
    long pos = blockIdx.x;
    half_t* p = d + pos * 256;
    int c = threadIdx.x;
    float v = (float)p[c];
    red[c] = v * v;
    __syncthreads();
    for (int s = 128; s > 0; s >>= 1) {
        if (c < s) red[c] += red[c + s];
        __syncthreads();
    }
    float nrm = fmaxf(sqrtf(red[0]), 1e-12f);
    p[c] = (half_t)(v / nrm);
}

__global__ __launch_bounds__(256)
void k_sample(const half_t* __restrict__ desc, const float* __restrict__ kpts,
              float* __restrict__ out)
{
    __shared__ float red[256];
    const int kp  = blockIdx.x;
    const int img = blockIdx.y;
    const float* K = kpts + ((long)img * 1024 + kp) * 2;
    float cx = (K[0] + 0.5f) / 8.f - 0.5f;
    float cy = (K[1] + 0.5f) / 8.f - 0.5f;
    cx = fminf(fmaxf(cx, 0.f), 63.f);
    cy = fminf(fmaxf(cy, 0.f), 63.f);
    int x0 = (int)floorf(cx), y0 = (int)floorf(cy);
    int x1 = min(x0 + 1, 63),  y1 = min(y0 + 1, 63);
    float fx = cx - (float)x0, fy = cy - (float)y0;

    const int c = threadIdx.x;
    const half_t* base = desc + (long)img * 64 * 64 * 256;
    float v00 = (float)base[((long)y0 * 64 + x0) * 256 + c];
    float v01 = (float)base[((long)y0 * 64 + x1) * 256 + c];
    float v10 = (float)base[((long)y1 * 64 + x0) * 256 + c];
    float v11 = (float)base[((long)y1 * 64 + x1) * 256 + c];
    float v = (1.f - fy) * ((1.f - fx) * v00 + fx * v01)
            +        fy  * ((1.f - fx) * v10 + fx * v11);

    red[c] = v * v;
    __syncthreads();
    for (int s = 128; s > 0; s >>= 1) {
        if (c < s) red[c] += red[c + s];
        __syncthreads();
    }
    float nrm = fmaxf(sqrtf(red[0]), 1e-12f);
    out[(((long)img * 1024) + kp) * 256 + c] = v / nrm;
}

// ===========================================================================
// Host side
// ===========================================================================
struct ConvP { const float *w, *scale, *bias, *mean, *var; };

// JAX pytree flatten order: dict keys sorted. d_in[0]=image, then per conv:
// bias, mean, scale, var, w. Conv order: backbone 0..7, desc0=8, desc1=9,
// det0=10, det1=11.
static ConvP getconv(void* const* d_in, int ci)
{
    int b = 1 + ci * 5;
    ConvP p;
    p.bias  = (const float*)d_in[b + 0];
    p.mean  = (const float*)d_in[b + 1];
    p.scale = (const float*)d_in[b + 2];
    p.var   = (const float*)d_in[b + 3];
    p.w     = (const float*)d_in[b + 4];
    return p;
}

static void conv(hipStream_t st, const half_t* in, const half_t* wp, half_t* out, ConvP p,
                 int N, int H, int W, int Cin, int Cout, int ks, int relu)
{
    int Ktot = ks * ks * Cin;
    int Kpad = (Ktot + 31) & ~31;
    long tiles = (long)N * H * (W >> 5);             // 32-pixel wave tiles
    dim3 grid((unsigned)((Cout + 15) >> 4), (unsigned)((tiles + 7) >> 3));
    k_conv_wmma<<<grid, 256, 0, st>>>(in, wp, p.scale, p.bias, p.mean, p.var,
                                      out, N, H, W, Cin, Cout, ks, relu, Kpad);
}

static inline unsigned blks(long n, int t) { return (unsigned)((n + t - 1) / t); }

extern "C" void kernel_launch(void* const* d_in, const int* in_sizes, int n_in,
                              void* d_out, int out_size, void* d_ws, size_t ws_size,
                              hipStream_t stream)
{
    (void)in_sizes; (void)n_in; (void)out_size; (void)ws_size;
    const int N = 4;
    const float* image = (const float*)d_in[0];

    char* ws = (char*)d_ws;
    size_t o = 0;
    auto carve = [&](size_t bytes) -> char* {
        char* p = ws + o;
        o = (o + bytes + 255) & ~(size_t)255;
        return p;
    };
    half_t* img16 = (half_t*)carve((size_t)N * 512 * 512 * 2);
    half_t* bufA  = (half_t*)carve((size_t)N * 512 * 512 * 64 * 2);
    half_t* bufB  = (half_t*)carve((size_t)N * 512 * 512 * 64 * 2);
    half_t* feat  = (half_t*)carve((size_t)N * 64 * 64 * 128 * 2);
    half_t* t256  = (half_t*)carve((size_t)N * 64 * 64 * 256 * 2);
    half_t* s65   = (half_t*)carve((size_t)N * 64 * 64 * 65  * 2);
    half_t* descb = (half_t*)carve((size_t)N * 64 * 64 * 256 * 2);
    float*  heat  = (float*)carve((size_t)N * 512 * 512 * 4);
    float*  mmb   = (float*)carve((size_t)N * 512 * 512 * 4);
    float*  t1f   = (float*)carve((size_t)N * 512 * 512 * 4);
    float*  t2f   = (float*)carve((size_t)N * 512 * 512 * 4);
    float*  t3f   = (float*)carve((size_t)N * 512 * 512 * 4);

    // ---- weight prep (f32 OIHW -> f16 [CoutPad][Kpad]) ----
    struct CDesc { int Cin, Cout, ks; };
    const CDesc cd[12] = {
        {1, 64, 3}, {64, 64, 3}, {64, 64, 3}, {64, 64, 3},
        {64, 128, 3}, {128, 128, 3}, {128, 128, 3}, {128, 128, 3},
        {128, 256, 3}, {256, 256, 1}, {128, 256, 3}, {256, 65, 1}
    };
    half_t* wp[12];
    for (int i = 0; i < 12; ++i) {
        int Ktot = cd[i].ks * cd[i].ks * cd[i].Cin;
        int Kpad = (Ktot + 31) & ~31;
        int CoutPad = (cd[i].Cout + 15) & ~15;
        long tot = (long)CoutPad * Kpad;
        wp[i] = (half_t*)carve((size_t)tot * 2);
        k_prep_w<<<blks(tot, 256), 256, 0, stream>>>(getconv(d_in, i).w, wp[i],
                                                     cd[i].Cin, cd[i].Cout, cd[i].ks,
                                                     CoutPad, Kpad);
    }

    float* out_kpts = (float*)d_out;                   // [4][1024][2]
    float* out_vals = out_kpts + (size_t)N * 1024 * 2; // [4][1024]
    float* out_desc = out_vals + (size_t)N * 1024;     // [4][1024][256]

    const long P512 = (long)N * 512 * 512;

    // ---- backbone ----
    k_cvt_img<<<blks(P512, 256), 256, 0, stream>>>(image, img16, P512);
    conv(stream, img16, wp[0], bufA, getconv(d_in, 0), N, 512, 512,   1,  64, 3, 1);
    conv(stream, bufA,  wp[1], bufB, getconv(d_in, 1), N, 512, 512,  64,  64, 3, 1);
    k_maxpool2<<<blks((long)N*256*256*64, 256), 256, 0, stream>>>(bufB, bufA, N, 512, 512, 64);
    conv(stream, bufA,  wp[2], bufB, getconv(d_in, 2), N, 256, 256,  64,  64, 3, 1);
    conv(stream, bufB,  wp[3], bufA, getconv(d_in, 3), N, 256, 256,  64,  64, 3, 1);
    k_maxpool2<<<blks((long)N*128*128*64, 256), 256, 0, stream>>>(bufA, bufB, N, 256, 256, 64);
    conv(stream, bufB,  wp[4], bufA, getconv(d_in, 4), N, 128, 128,  64, 128, 3, 1);
    conv(stream, bufA,  wp[5], bufB, getconv(d_in, 5), N, 128, 128, 128, 128, 3, 1);
    k_maxpool2<<<blks((long)N*64*64*128, 256), 256, 0, stream>>>(bufB, bufA, N, 128, 128, 128);
    conv(stream, bufA,  wp[6], bufB, getconv(d_in, 6), N,  64,  64, 128, 128, 3, 1);
    conv(stream, bufB,  wp[7], feat, getconv(d_in, 7), N,  64,  64, 128, 128, 3, 1);

    // ---- detector head ----
    conv(stream, feat, wp[10], t256, getconv(d_in, 10), N, 64, 64, 128, 256, 3, 1);
    conv(stream, t256, wp[11], s65,  getconv(d_in, 11), N, 64, 64, 256,  65, 1, 0);
    k_softmax_shuffle<<<blks(N*64*64, 256), 256, 0, stream>>>(s65, heat, N);

    // ---- simple_nms (radius 4, 2 iterations) ----
    k_pool9<<<blks(P512, 256), 256, 0, stream>>>(heat, t1f, N);
    k_eq<<<blks(P512, 256), 256, 0, stream>>>(heat, t1f, mmb, P512);
    for (int it = 0; it < 2; ++it) {
        k_pool9<<<blks(P512, 256), 256, 0, stream>>>(mmb, t1f, N);
        k_supp<<<blks(P512, 256), 256, 0, stream>>>(t1f, heat, t2f, t3f, P512);
        k_pool9<<<blks(P512, 256), 256, 0, stream>>>(t2f, t1f, N);
        k_upd<<<blks(P512, 256), 256, 0, stream>>>(mmb, t2f, t1f, t3f, P512);
    }
    k_fin<<<blks(P512, 256), 256, 0, stream>>>(mmb, heat, t2f, N);

    // ---- top-1024 per image ----
    k_topk<<<N, 1024, 0, stream>>>(t2f, out_kpts, out_vals);

    // ---- descriptor head + sampling ----
    conv(stream, feat, wp[8], t256,  getconv(d_in, 8), N, 64, 64, 128, 256, 3, 1);
    conv(stream, t256, wp[9], descb, getconv(d_in, 9), N, 64, 64, 256, 256, 1, 0);
    k_descnorm<<<N * 64 * 64, 256, 0, stream>>>(descb);
    k_sample<<<dim3(1024, N), 256, 0, stream>>>(descb, out_kpts, out_desc);
}
